// MultiHeadAttention_67757404062370
// MI455X (gfx1250) — compile-verified, matrix operand order checked by reading, not run on hardware
//
#include <hip/hip_runtime.h>
#include <hip/hip_bf16.h>

// ---------- types ----------
typedef __attribute__((ext_vector_type(16))) _Float16 v16h;
typedef __attribute__((ext_vector_type(8)))  float    v8f;
typedef __attribute__((ext_vector_type(4)))  unsigned int v4u;
typedef __attribute__((ext_vector_type(2)))  unsigned int v2u;

union Frag16 { v16h h; v4u u[2]; };

// Load a 16-half WMMA fragment (A or B operand, 16x32 f16 layout).
// Per CDNA5 ISA: lanes 0-15 hold K in [0,8)+[16,24), lanes 16-31 hold
// K in [8,16)+[24,32); ascending-K pairs pack into consecutive VGPRs,
// so the fragment is exactly two contiguous 16-byte chunks per lane.
__device__ inline v16h frag_ld(const _Float16* rowbase, int half) {
  Frag16 f;
  const _Float16* p = rowbase + (half ? 8 : 0);
  f.u[0] = *(const v4u*)(p);
  f.u[1] = *(const v4u*)(p + 16);
  return f.h;
}

__device__ inline v8f wmma_f16(v16h a, v16h b, v8f c) {
  return __builtin_amdgcn_wmma_f32_16x16x32_f16(
      /*neg_a=*/false, a, /*neg_b=*/false, b,
      /*c_mod=*/(short)0, c, /*reuse_a=*/false, /*reuse_b=*/false);
}

// ---------- GEMM with bias: C[M,1024] = A[M,1024] @ W[1024,1024] + b ----------
// 128x64 block tile, BK=32, double-buffered LDS staging (one barrier/step).
// 8 waves: wave w owns row tile w (16 rows) x all four 16-col tiles:
// per k-step 1 A-fragment is reused across 4 WMMAs.
template <typename AT>
__global__ __launch_bounds__(256) void proj_gemm(
    const AT* __restrict__ A, const float* __restrict__ W,
    const float* __restrict__ bias, float* __restrict__ Cf,
    _Float16* __restrict__ Ch, int M) {
  constexpr int K = 1024, N = 1024;
  __shared__ _Float16 As[2][128 * 32];  // A tiles, row-major (ld=32): 2 x 8 KB
  __shared__ _Float16 Wt[2][64 * 32];   // W tiles transposed Wt[n][k]: 2 x 4 KB

  const int tid  = threadIdx.x;
  const int wave = tid >> 5, lane = tid & 31;
  const int half = lane >> 4, l16 = lane & 15;
  const int m0g = blockIdx.x * 128;
  const int n0g = blockIdx.y * 64;

  v8f acc[4] = {{}, {}, {}, {}};

  // Cooperative staging of one 128x32 A tile + 32x64 W tile (f16, W transposed).
  auto stage = [&](int buf, int k0) {
    {  // A: each thread converts 16 contiguous elements of one row
      int r = tid >> 1;
      int c = (tid & 1) * 16;
      int gm = m0g + r;
      alignas(16) _Float16 tmp[16];
      if (gm < M) {
        const AT* src = A + (size_t)gm * K + k0 + c;
        __builtin_prefetch(src + 64, 0, 0);  // two k-steps ahead
#pragma unroll
        for (int i = 0; i < 16; ++i) tmp[i] = (_Float16)src[i];
      } else {
#pragma unroll
        for (int i = 0; i < 16; ++i) tmp[i] = (_Float16)0.0f;
      }
      *(v4u*)&As[buf][r * 32 + c]     = *(v4u*)&tmp[0];
      *(v4u*)&As[buf][r * 32 + c + 8] = *(v4u*)&tmp[8];
    }
    {  // W: 32x64 -> Wt[n][k]
      int k = tid >> 3;
      int n = (tid & 7) * 8;
      const float* src = W + (size_t)(k0 + k) * N + n0g + n;
      __builtin_prefetch(src + (size_t)64 * N, 0, 0);
#pragma unroll
      for (int i = 0; i < 8; ++i) Wt[buf][(n + i) * 32 + k] = (_Float16)src[i];
    }
  };

  stage(0, 0);
  __syncthreads();

  for (int k0 = 0; k0 < K; k0 += 32) {
    const int buf = (k0 >> 5) & 1;
    if (k0 + 32 < K) stage(buf ^ 1, k0 + 32);  // overlap next stage with WMMAs

    v16h a = frag_ld(&As[buf][(wave * 16 + l16) * 32], half);
#pragma unroll
    for (int t = 0; t < 4; ++t) {
      v16h bb = frag_ld(&Wt[buf][(t * 16 + l16) * 32], half);
      acc[t] = wmma_f16(a, bb, acc[t]);
    }
    __syncthreads();
  }

  // Epilogue: C layout per ISA -> lane holds column l16, rows r + 8*half.
#pragma unroll
  for (int t = 0; t < 4; ++t) {
    const int n = n0g + t * 16 + l16;
    const float bia = bias[n];
#pragma unroll
    for (int r = 0; r < 8; ++r) {
      int m = m0g + wave * 16 + half * 8 + r;
      if (m < M) {
        float v = acc[t][r] + bia;
        if (Cf) Cf[(size_t)m * N + n] = v;
        if (Ch) Ch[(size_t)m * N + n] = (_Float16)v;
      }
    }
  }
}

// ---------- fused attention core ----------
// One block per (b, h, 16 query rows). 8 waves.
// Phase 1: scores (16x1024) = q_tile @ K^T via WMMA  (wave w -> k cols [128w,128w+128))
// Phase 2: fp32 softmax with relu/scale/mask, att_weights written coalesced to d_out
// Phase 3: out (16x64) = att @ V via WMMA, split-K over two wave groups
__global__ __launch_bounds__(256) void attn_kernel(
    const _Float16* __restrict__ Q, const _Float16* __restrict__ Kc,
    const _Float16* __restrict__ V, const int* __restrict__ mask,
    float* __restrict__ attw, _Float16* __restrict__ O16) {
  constexpr int D = 1024, S = 1024, SQ = 1025, DH = 64, H = 16;
  __shared__ _Float16 qs[16 * 64];        // 2 KB
  __shared__ _Float16 sc[16 * 1024];      // 32 KB (scores -> att, f16)
  __shared__ _Float16 vt[2 * 64 * 32];    // 8 KB (transposed V tiles, per group)
  __shared__ float    oacc[2 * 16 * 64];  // 8 KB (split-K partials)

  const int tid  = threadIdx.x;
  const int wave = tid >> 5, lane = tid & 31;
  const int half = lane >> 4, l16 = lane & 15;
  const int bid = blockIdx.x;
  const int b  = bid >> 10;
  const int h  = (bid >> 6) & 15;
  const int q0 = (bid & 63) * 16;

  // ---- load q tile (rows q0+1 .. q0+16 of the 1025: the [1:] slice) ----
  {
    int qi = tid >> 4;
    int d  = (tid & 15) * 4;
    const _Float16* src = Q + ((size_t)(b * SQ + q0 + qi + 1)) * D + h * DH + d;
    *(v2u*)&qs[qi * 64 + d] = *(const v2u*)src;
  }
  __syncthreads();

  // ---- phase 1: scores ----
  for (int kt = 0; kt < 8; ++kt) {
    const int kbase = wave * 128 + kt * 16;
    v8f acc = {};
    const int krow = kbase + l16;  // k cols 0..1023 (the [:-1] slice)
    const _Float16* kp = Kc + ((size_t)(b * SQ + krow)) * D + h * DH;
#pragma unroll
    for (int d0 = 0; d0 < 64; d0 += 32) {
      v16h a  = frag_ld(&qs[l16 * 64 + d0], half);
      v16h bb = frag_ld(kp + d0, half);
      acc = wmma_f16(a, bb, acc);
    }
#pragma unroll
    for (int r = 0; r < 8; ++r) {
      int m = half * 8 + r;
      sc[m * 1024 + kbase + l16] = (_Float16)acc[r];
    }
  }
  __syncthreads();

  // ---- phase 2: softmax (wave handles 2 rows; lane strides columns) ----
  for (int rr = 0; rr < 2; ++rr) {
    const int m = wave * 2 + rr;
    const int* mrow = mask + ((size_t)b * S + q0 + m) * S;
    float vals[32];
    float mx = -3.0e4f;
#pragma unroll
    for (int j = 0; j < 32; ++j) {
      int c = lane + j * 32;
      float raw = (float)sc[m * 1024 + c];
      float v = fmaxf(raw * 0.125f, 0.0f) + (float)mrow[c] * -30000.0f;
      vals[j] = v;
      mx = fmaxf(mx, v);
    }
#pragma unroll
    for (int off = 16; off > 0; off >>= 1) mx = fmaxf(mx, __shfl_xor(mx, off));
    float s = 0.0f;
#pragma unroll
    for (int j = 0; j < 32; ++j) {
      vals[j] = __expf(vals[j] - mx);
      s += vals[j];
    }
#pragma unroll
    for (int off = 16; off > 0; off >>= 1) s += __shfl_xor(s, off);
    const float inv = 1.0f / s;
    float* arow = attw + (((size_t)(b * H + h) * 1024) + q0 + m) * 1024;
#pragma unroll
    for (int j = 0; j < 32; ++j) {
      int c = lane + j * 32;
      float e = vals[j] * inv;
      arow[c] = e;                       // coalesced fp32 store (dominant HBM cost)
      sc[m * 1024 + c] = (_Float16)e;    // f16 att for the WMMA stage
    }
  }
  __syncthreads();

  // ---- phase 3: out = att @ V (split-K: group g covers k in [512g, 512g+512)) ----
  const int ct = wave & 3;    // 16-col tile of the 64-wide head
  const int g  = wave >> 2;   // split-K group
  const int n0 = ct * 16;
  v8f acc = {};
  for (int it = 0; it < 16; ++it) {
    const int k0 = g * 512 + it * 32;
    {  // group-cooperative transpose-stage of V[k0..k0+32)[64] -> vt[g][n][k]
      int tt = tid & 127;
      int k = tt >> 2;
      int n = (tt & 3) * 16;
      const _Float16* src = V + ((size_t)(b * S + k0 + k)) * D + h * DH + n;
#pragma unroll
      for (int i = 0; i < 16; ++i) vt[g * 2048 + (n + i) * 32 + k] = src[i];
    }
    __syncthreads();
    v16h a  = frag_ld(&sc[l16 * 1024 + k0], half);
    v16h bb = frag_ld(&vt[g * 2048 + (n0 + l16) * 32], half);
    acc = wmma_f16(a, bb, acc);
    __syncthreads();
  }
#pragma unroll
  for (int r = 0; r < 8; ++r) {
    int m = half * 8 + r;
    oacc[g * 1024 + m * 64 + n0 + l16] = acc[r];
  }
  __syncthreads();

  // ---- reduce split-K partials, emit f16 head output in [B,S,D] layout ----
  {
    int i = tid;
#pragma unroll
    for (int rep = 0; rep < 4; ++rep, i += 256) {
      int m = i >> 6, n = i & 63;
      float v = oacc[i] + oacc[1024 + i];
      O16[((size_t)(b * S + q0 + m)) * D + h * DH + n] = (_Float16)v;
    }
  }
}

// ---------- host launcher ----------
extern "C" void kernel_launch(void* const* d_in, const int* in_sizes, int n_in,
                              void* d_out, int out_size, void* d_ws, size_t ws_size,
                              hipStream_t stream) {
  (void)in_sizes; (void)n_in; (void)out_size; (void)ws_size;
  const float* v  = (const float*)d_in[0];
  const float* k  = (const float*)d_in[1];
  const float* q  = (const float*)d_in[2];
  const int*   mk = (const int*)d_in[3];
  const float* Wq = (const float*)d_in[4];
  const float* bq = (const float*)d_in[5];
  const float* Wk = (const float*)d_in[6];
  const float* bk = (const float*)d_in[7];
  const float* Wv = (const float*)d_in[8];
  const float* bv = (const float*)d_in[9];
  const float* Wo = (const float*)d_in[10];
  const float* bo = (const float*)d_in[11];

  float* out  = (float*)d_out;                          // (2,1024,1024) fp32
  float* attw = out + (size_t)2 * 1024 * 1024;          // (2,16,1024,1024) fp32

  _Float16* q16 = (_Float16*)d_ws;                      // (2,1025,1024)
  _Float16* k16 = q16 + (size_t)2 * 1025 * 1024;        // (2,1025,1024)
  _Float16* v16 = k16 + (size_t)2 * 1025 * 1024;        // (2,1024,1024)
  _Float16* o16 = v16 + (size_t)2 * 1024 * 1024;        // (2,1024,1024)

  dim3 blk(256);
  dim3 g_qk((2050 + 127) / 128, 16);
  dim3 g_sv((2048 + 127) / 128, 16);

  proj_gemm<float><<<g_qk, blk, 0, stream>>>(q, Wq, bq, nullptr, q16, 2050);
  proj_gemm<float><<<g_qk, blk, 0, stream>>>(k, Wk, bk, nullptr, k16, 2050);
  proj_gemm<float><<<g_sv, blk, 0, stream>>>(v, Wv, bv, nullptr, v16, 2048);

  attn_kernel<<<dim3(2 * 16 * 64), blk, 0, stream>>>(q16, k16, v16, mk, attw, o16);

  proj_gemm<_Float16><<<g_sv, blk, 0, stream>>>(o16, Wo, bo, out, nullptr, 2048);
}
